// Conv_56994216017979
// MI455X (gfx1250) — matrix operand order repaired and checked by reading, not compiled, not run on hardware
//
#include <hip/hip_runtime.h>
#include <hip/hip_bf16.h>
#include <math.h>

typedef __attribute__((ext_vector_type(16))) _Float16 v16h;
typedef __attribute__((ext_vector_type(8)))  float    v8f;

#define GN1 256
#define GN2 128
#define GNE 4096

// ---------------------------------------------------------------------------
// Weight pre-pack: OIHW f32 -> f16 fragments in contiguous-k A-fragment order (B loads use the same).
// Packed layout: [co_blk][ky*3+kx][cb][lane(32)][e(16)] contiguous halves.
// A 16x32 f16 fragment mapping: lane l: M = l%16, K(e) = 16*(l/16) + e
// ---------------------------------------------------------------------------
template<int CI, int CIP, int CO>
__global__ void pack_w(const float* __restrict__ Wq, _Float16* __restrict__ P)
{
  constexpr int KB = CIP / 32;
  int j = blockIdx.x * blockDim.x + threadIdx.x;
  if (j >= (CO / 16) * 9 * KB * 512) return;
  const int e    = j & 15;
  const int lane = (j >> 4) & 31;
  const int cb   = (j >> 9) % KB;
  const int t    = j / (512 * KB);
  const int kk   = t % 9, cob = t / 9;
  const int ky = kk / 3, kx = kk % 3;
  const int g = lane >> 4, ln = lane & 15;
  const int m  = cob * 16 + ln;
  const int k  = (g << 4) + e;
  const int ci = cb * 32 + k;
  float v = (ci < CI) ? Wq[((m * CI + ci) * 3 + ky) * 3 + kx] : 0.0f;
  P[j] = (_Float16)v;
}

// f32 NCHW [5][512][512] -> f16 halo NHWC [514][514][32] (c>=5 stays zero)
__global__ void cvt_in(const float* __restrict__ X, _Float16* __restrict__ Y)
{
  int i = blockIdx.x * blockDim.x + threadIdx.x;
  if (i >= 5 * 512 * 512) return;
  const int c = i / (512 * 512);
  const int r = i % (512 * 512);
  const int y = r >> 9, x = r & 511;
  Y[((size_t)(y + 1) * 514 + (x + 1)) * 32 + c] = (_Float16)X[i];
}

// 2x2 maxpool of f32 NCHW conv output -> f16 halo NHWC input of next layer
template<int C, int H, int W>
__global__ void pool_cvt_nhwc(const float* __restrict__ X, _Float16* __restrict__ Y)
{
  constexpr int HO = H / 2, WO = W / 2;
  int i = blockIdx.x * blockDim.x + threadIdx.x;
  if (i >= HO * WO * C) return;
  const int c = i % C;
  const int x = (i / C) % WO;
  const int y = i / (C * WO);
  const float* p = X + ((size_t)c * H + 2 * y) * W + 2 * x;
  float v = fmaxf(fmaxf(p[0], p[1]), fmaxf(p[W], p[W + 1]));
  Y[((size_t)(y + 1) * (WO + 2) + (x + 1)) * C + c] = (_Float16)v;
}

// ---------------------------------------------------------------------------
// Conv 3x3 SAME + bias + ReLU via WMMA implicit GEMM, branch-free operands.
// One wave computes 16(co) x 16(x) outputs at row y. Input: f16 halo NHWC.
// Per (tap, cb): A = 32B contiguous load (packed), B = 32B contiguous load.
// ---------------------------------------------------------------------------
template<int CIP, int CO, int H, int W>
__global__ __launch_bounds__(32)
void conv3x3_relu_wmma(const _Float16* __restrict__ I, const _Float16* __restrict__ WP,
                       const float* __restrict__ Bs, float* __restrict__ Y)
{
  const int lane = threadIdx.x & 31;
  const int g = lane >> 4, ln = lane & 15;
  const int x0 = blockIdx.x << 4;
  const int y  = blockIdx.y;
  const int cob = blockIdx.z;
  constexpr int KB  = CIP / 32;
  constexpr int WH  = W + 2;

  const _Float16* wbase = WP + ((size_t)cob * 9 * KB) * 512 + lane * 16;
  v8f acc = {};
  #pragma unroll
  for (int ky = 0; ky < 3; ++ky) {
    if (ky < 2)  // prefetch next tap row (global_prefetch_b8)
      __builtin_prefetch(I + ((size_t)(y + ky + 1) * WH + (x0 + ln)) * CIP, 0, 1);
    #pragma unroll
    for (int kx = 0; kx < 3; ++kx) {
      const _Float16* ibase =
          I + ((size_t)(y + ky) * WH + (x0 + ln + kx)) * CIP + (g << 4);
      const _Float16* wb = wbase + (size_t)(ky * 3 + kx) * KB * 512;
      #pragma unroll
      for (int cb = 0; cb < KB; ++cb) {
        v16h a = *(const v16h*)(wb + (size_t)cb * 512);
        v16h b = *(const v16h*)(ibase + cb * 32);
        acc = __builtin_amdgcn_wmma_f32_16x16x32_f16(
            false, a, false, b, (short)0, acc, false, false);
      }
    }
  }
  #pragma unroll
  for (int r = 0; r < 8; ++r) {
    const int co = (cob << 4) + r + (g << 3);
    float v = acc[r] + Bs[co];
    Y[((size_t)co * H + y) * W + (x0 + ln)] = v > 0.0f ? v : 0.0f;
  }
}

// plain f32 maxpool for the last layer (NCHW -> NCHW == node features)
template<int C, int H, int W>
__global__ void maxpool2x2(const float* __restrict__ X, float* __restrict__ Y)
{
  const int HO = H / 2, WO = W / 2;
  int i = blockIdx.x * blockDim.x + threadIdx.x;
  if (i >= C * HO * WO) return;
  const int x = i % WO, y = (i / WO) % HO, c = i / (WO * HO);
  const float* p = X + ((size_t)c * H + 2 * y) * W + 2 * x;
  Y[i] = fmaxf(fmaxf(p[0], p[1]), fmaxf(p[W], p[W + 1]));
}

__global__ void zerof(float* p, int n)
{
  int i = blockIdx.x * blockDim.x + threadIdx.x;
  if (i < n) p[i] = 0.0f;
}

// ---------------------------------------------------------------------------
// SAGE / diff-pool pieces (tiny; VALU kernels)
// ---------------------------------------------------------------------------
__global__ void sage_agg(const float* __restrict__ X, const int* __restrict__ src,
                         const int* __restrict__ dst, int E, int F,
                         float* __restrict__ agg, float* __restrict__ cnt)
{
  int i = blockIdx.x * blockDim.x + threadIdx.x;
  int e = i / F, f = i % F;
  if (e >= E) return;
  atomicAdd(&agg[dst[e] * F + f], X[src[e] * F + f]);
  if (f == 0) atomicAdd(&cnt[dst[e]], 1.0f);
}

__global__ void sage3_agg(const float* __restrict__ XP, const int* __restrict__ cols,
                          int N, int F, float* __restrict__ agg, float* __restrict__ cnt)
{
  int i = blockIdx.x * blockDim.x + threadIdx.x;
  int e = i / F, f = i % F;
  if (e >= N) return;
  atomicAdd(&agg[cols[e] * F + f], XP[e * F + f]);
  if (f == 0) atomicAdd(&cnt[cols[e]], 1.0f);
}

__global__ void sage_lin(const float* __restrict__ X, const float* __restrict__ agg,
                         const float* __restrict__ cnt, const float* __restrict__ wl,
                         const float* __restrict__ bl, const float* __restrict__ wr,
                         int N, int CI, int CO, float* __restrict__ out)
{
  int i = blockIdx.x * blockDim.x + threadIdx.x;
  if (i >= N * CO) return;
  int n = i / CO, co = i % CO;
  float inv = 1.0f / fmaxf(cnt[n], 1.0f);
  float s = bl[co];
  for (int ci = 0; ci < CI; ++ci)
    s += agg[n * CI + ci] * inv * wl[co * CI + ci] + X[n * CI + ci] * wr[co * CI + ci];
  out[i] = s;
}

__global__ void rownorm(float* __restrict__ out, int N, int CO, int do_tanh)
{
  int n = blockIdx.x * blockDim.x + threadIdx.x;
  if (n >= N) return;
  float ss = 0.0f;
  for (int c = 0; c < CO; ++c) { float v = out[n * CO + c]; ss += v * v; }
  float inv = 1.0f / fmaxf(sqrtf(ss), 1e-12f);
  for (int c = 0; c < CO; ++c) {
    float v = out[n * CO + c] * inv;
    out[n * CO + c] = do_tanh ? tanhf(v) : v;
  }
}

__global__ void build_adj(const int* __restrict__ src, const int* __restrict__ dst,
                          int E, int N, float* __restrict__ adj)
{
  int e = blockIdx.x * blockDim.x + threadIdx.x;
  if (e >= E) return;
  atomicAdd(&adj[src[e] * N + dst[e]], 1.0f);
}

__global__ void softmax_ent(float* __restrict__ S, int N, int C, float* __restrict__ ent)
{
  int n = blockIdx.x * blockDim.x + threadIdx.x;
  if (n >= N) return;
  float mx = -1e30f;
  for (int c = 0; c < C; ++c) mx = fmaxf(mx, S[n * C + c]);
  float sum = 0.0f;
  for (int c = 0; c < C; ++c) sum += expf(S[n * C + c] - mx);
  float inv = 1.0f / sum, en = 0.0f;
  for (int c = 0; c < C; ++c) {
    float p = expf(S[n * C + c] - mx) * inv;
    S[n * C + c] = p;
    en += -p * logf(p + 1e-15f);
  }
  atomicAdd(ent, en);
}

__global__ void st_x(const float* __restrict__ S, const float* __restrict__ X,
                     int N, int C, int F, float* __restrict__ OUT)
{
  int i = blockIdx.x * blockDim.x + threadIdx.x;
  if (i >= C * F) return;
  int c = i / F, f = i % F;
  float s = 0.0f;
  for (int n = 0; n < N; ++n) s += S[n * C + c] * X[n * F + f];
  OUT[i] = s;
}

__global__ void a_s(const float* __restrict__ A, const float* __restrict__ S,
                    int N, int C, float* __restrict__ T)
{
  int i = blockIdx.x * blockDim.x + threadIdx.x;
  if (i >= N * C) return;
  int n = i / C, d = i % C;
  float s = 0.0f;
  for (int m = 0; m < N; ++m) s += A[n * N + m] * S[m * C + d];
  T[i] = s;
}

__global__ void link_loss(const float* __restrict__ A, const float* __restrict__ S,
                          int N, int C, float* __restrict__ accum)
{
  int i = blockIdx.x * blockDim.x + threadIdx.x;
  float v = 0.0f;
  if (i < N * N) {
    int n = i / N, m = i % N;
    float d = A[i];
    for (int c = 0; c < C; ++c) d -= S[n * C + c] * S[m * C + c];
    v = d * d;
  }
  __shared__ float sh[256];
  sh[threadIdx.x] = v; __syncthreads();
  for (int s = 128; s > 0; s >>= 1) {
    if ((int)threadIdx.x < s) sh[threadIdx.x] += sh[threadIdx.x + s];
    __syncthreads();
  }
  if (threadIdx.x == 0) atomicAdd(accum, sh[0]);
}

__global__ void rowmax_bin(const float* __restrict__ ADJP, int N,
                           float* __restrict__ edge_out, int* __restrict__ cols,
                           float* __restrict__ out_ei2)
{
  int n = blockIdx.x * blockDim.x + threadIdx.x;
  if (n >= N) return;
  float mx = ADJP[n * N]; int am = 0;
  for (int j = 1; j < N; ++j) {
    float v = ADJP[n * N + j];
    if (v > mx) { mx = v; am = j; }
  }
  for (int j = 0; j < N; ++j)
    edge_out[n * N + j] = (ADJP[n * N + j] == mx) ? 1.0f : 0.0f;
  cols[n] = am;
  out_ei2[n]     = (float)n;
  out_ei2[N + n] = (float)am;
}

__global__ void finalize(const float* __restrict__ nodes, const float* __restrict__ scal,
                         float* __restrict__ out)
{
  float x0 = 0.0f, x1 = 0.0f;
  for (int n = 0; n < GN2; ++n) { x0 += nodes[n * 2]; x1 += nodes[n * 2 + 1]; }
  out[0] = x0; out[1] = x1;
  float ll1 = sqrtf(scal[0]) / (float)(GN1 * GN1);
  float ll2 = sqrtf((float)(GN2 * GN2 - GN2)) / (float)(GN2 * GN2);
  out[2] = ll1 + ll2;
  float el1 = scal[1] / (float)GN1;
  float el2 = -logf(1.0f + 1e-15f);   // == 0 in f32
  out[3] = el1 + el2;
}

// ---------------------------------------------------------------------------
// Workspace layout (float offsets)
// ---------------------------------------------------------------------------
static const size_t F_A     = 0;                       // 32*512*512 f32 conv-out
static const size_t F_HBUF  = F_A + 8388608;           // 12,519,424 halves
static const size_t HB_LEN_F = 6259712;                // = halves/2
static const size_t F_WPACK = F_HBUF + HB_LEN_F;       // 986,112 halves
static const size_t WP_LEN_F = 493056;
static const size_t F_ORG   = F_WPACK + WP_LEN_F;      // 65536
static const size_t F_X1    = F_ORG  + 65536;          // 32768
static const size_t F_S1    = F_X1   + 32768;          // 32768
static const size_t F_T     = F_S1   + 32768;          // 32768
static const size_t F_ADJP  = F_T    + 32768;          // 16384
static const size_t F_XP    = F_ADJP + 16384;          // 16384
static const size_t F_COLS  = F_XP   + 16384;          // 128 ints
static const size_t F_ZERO  = F_COLS + 128;            // --- zeroed each call ---
static const size_t F_AGG   = F_ZERO;                  // 65536
static const size_t F_CNT   = F_AGG  + 65536;          // 256
static const size_t F_ADJ1  = F_CNT  + 256;            // 65536
static const size_t F_AGG3  = F_ADJ1 + 65536;          // 16384
static const size_t F_CNT3  = F_AGG3 + 16384;          // 128
static const size_t F_SCAL  = F_CNT3 + 128;            // 8
static const size_t ZERO_LEN = (F_SCAL + 8) - F_ZERO;

// half-offsets inside HBUF / WPACK
static const size_t H_I1 = 0;          // 514*514*32  = 8,454,272
static const size_t H_I2 = 8454272;    // 258*258*32  = 2,130,048
static const size_t H_I3 = 10584320;   // 130*130*64  = 1,081,600
static const size_t H_I4 = 11665920;   // 66*66*128   =   557,568
static const size_t H_I5 = 12223488;   // 34*34*256   =   295,936
static const size_t H_W1 = 0;          // 9216
static const size_t H_W2 = 9216;       // 18432
static const size_t H_W3 = 27648;      // 73728
static const size_t H_W4 = 101376;     // 294912
static const size_t H_W5 = 396288;     // 589824

extern "C" void kernel_launch(void* const* d_in, const int* in_sizes, int n_in,
                              void* d_out, int out_size, void* d_ws, size_t ws_size,
                              hipStream_t stream)
{
  (void)in_sizes; (void)n_in; (void)out_size; (void)ws_size;
  const float* inp = (const float*)d_in[0];
  const int*   ei  = (const int*)d_in[1];
  const int *src = ei, *dst = ei + GNE;
  const float *w1 = (const float*)d_in[2],  *b1 = (const float*)d_in[3];
  const float *w2 = (const float*)d_in[4],  *b2 = (const float*)d_in[5];
  const float *w3 = (const float*)d_in[6],  *b3 = (const float*)d_in[7];
  const float *w4 = (const float*)d_in[8],  *b4 = (const float*)d_in[9];
  const float *w5 = (const float*)d_in[10], *b5 = (const float*)d_in[11];
  const float *s1wl = (const float*)d_in[12], *s1bl = (const float*)d_in[13], *s1wr = (const float*)d_in[14];
  const float *s2wl = (const float*)d_in[15], *s2bl = (const float*)d_in[16], *s2wr = (const float*)d_in[17];
  const float *s3wl = (const float*)d_in[18], *s3bl = (const float*)d_in[19], *s3wr = (const float*)d_in[20];

  float* ws = (float*)d_ws;
  float* A  = ws + F_A;
  _Float16* HB  = (_Float16*)(ws + F_HBUF);
  _Float16* I1 = HB + H_I1, *I2 = HB + H_I2, *I3 = HB + H_I3, *I4 = HB + H_I4, *I5 = HB + H_I5;
  _Float16* WPK = (_Float16*)(ws + F_WPACK);
  _Float16* W1P = WPK + H_W1, *W2P = WPK + H_W2, *W3P = WPK + H_W3, *W4P = WPK + H_W4, *W5P = WPK + H_W5;
  float* ORG  = ws + F_ORG;
  float* X1   = ws + F_X1;
  float* S1   = ws + F_S1;
  float* T    = ws + F_T;
  float* ADJP = ws + F_ADJP;
  float* XP   = ws + F_XP;
  int*   COLS = (int*)(ws + F_COLS);
  float* AGG  = ws + F_AGG;
  float* CNT  = ws + F_CNT;
  float* ADJ1 = ws + F_ADJ1;
  float* AGG3 = ws + F_AGG3;
  float* CNT3 = ws + F_CNT3;
  float* SCAL = ws + F_SCAL;

  float* out       = (float*)d_out;
  float* out_nodes = out + 4;
  float* out_edge  = out + 260;
  float* out_ei2   = out + 16644;

  // ---- zero f16 staging (halos + channel pad) and atomic scratch ----
  zerof<<<((int)HB_LEN_F + 255) / 256, 256, 0, stream>>>(ws + F_HBUF, (int)HB_LEN_F);
  zerof<<<((int)ZERO_LEN + 255) / 256, 256, 0, stream>>>(ws + F_ZERO, (int)ZERO_LEN);

  // ---- weight pre-pack into WMMA A-fragment order ----
  pack_w<5,  32, 32 ><<<(9216   + 255) / 256, 256, 0, stream>>>(w1, W1P);
  pack_w<32, 32, 64 ><<<(18432  + 255) / 256, 256, 0, stream>>>(w2, W2P);
  pack_w<64, 64, 128><<<(73728  + 255) / 256, 256, 0, stream>>>(w3, W3P);
  pack_w<128,128,256><<<(294912 + 255) / 256, 256, 0, stream>>>(w4, W4P);
  pack_w<256,256,256><<<(589824 + 255) / 256, 256, 0, stream>>>(w5, W5P);

  // ---- CNN: cvt -> [conv -> pool+cvt]x4 -> conv -> pool ----
  cvt_in<<<(5 * 512 * 512 + 255) / 256, 256, 0, stream>>>(inp, I1);
  conv3x3_relu_wmma<32, 32, 512, 512><<<dim3(32, 512, 2),  32, 0, stream>>>(I1, W1P, b1, A);
  pool_cvt_nhwc<32, 512, 512><<<(32 * 256 * 256 + 255) / 256, 256, 0, stream>>>(A, I2);
  conv3x3_relu_wmma<32, 64, 256, 256><<<dim3(16, 256, 4),  32, 0, stream>>>(I2, W2P, b2, A);
  pool_cvt_nhwc<64, 256, 256><<<(64 * 128 * 128 + 255) / 256, 256, 0, stream>>>(A, I3);
  conv3x3_relu_wmma<64, 128, 128, 128><<<dim3(8, 128, 8),  32, 0, stream>>>(I3, W3P, b3, A);
  pool_cvt_nhwc<128, 128, 128><<<(128 * 64 * 64 + 255) / 256, 256, 0, stream>>>(A, I4);
  conv3x3_relu_wmma<128, 256, 64, 64><<<dim3(4, 64, 16),   32, 0, stream>>>(I4, W4P, b4, A);
  pool_cvt_nhwc<256, 64, 64><<<(256 * 32 * 32 + 255) / 256, 256, 0, stream>>>(A, I5);
  conv3x3_relu_wmma<256, 256, 32, 32><<<dim3(2, 32, 16),   32, 0, stream>>>(I5, W5P, b5, A);
  maxpool2x2<256, 32, 32><<<(256 * 16 * 16 + 255) / 256,   256, 0, stream>>>(A, ORG);

  // ---- SAGE 1 & 2 on ORG (256 nodes x 256 feats) ----
  sage_agg<<<(GNE * 256 + 255) / 256, 256, 0, stream>>>(ORG, src, dst, GNE, 256, AGG, CNT);
  sage_lin<<<(GN1 * 128 + 255) / 256, 256, 0, stream>>>(ORG, AGG, CNT, s1wl, s1bl, s1wr, GN1, 256, 128, X1);
  rownorm<<<(GN1 + 127) / 128, 128, 0, stream>>>(X1, GN1, 128, 0);
  sage_lin<<<(GN1 * 128 + 255) / 256, 256, 0, stream>>>(ORG, AGG, CNT, s2wl, s2bl, s2wr, GN1, 256, 128, S1);
  rownorm<<<(GN1 + 127) / 128, 128, 0, stream>>>(S1, GN1, 128, 0);

  // ---- dense adjacency + diff_pool #1 ----
  build_adj<<<(GNE + 255) / 256, 256, 0, stream>>>(src, dst, GNE, GN1, ADJ1);
  softmax_ent<<<(GN1 + 127) / 128, 128, 0, stream>>>(S1, GN1, 128, &SCAL[1]);
  st_x<<<(128 * 128 + 255) / 256, 256, 0, stream>>>(S1, X1, GN1, 128, 128, XP);
  a_s<<<(GN1 * 128 + 255) / 256, 256, 0, stream>>>(ADJ1, S1, GN1, 128, T);
  st_x<<<(128 * 128 + 255) / 256, 256, 0, stream>>>(S1, T, GN1, 128, 128, ADJP);
  link_loss<<<(GN1 * GN1 + 255) / 256, 256, 0, stream>>>(ADJ1, S1, GN1, 128, &SCAL[0]);

  // ---- binarize / argmax -> edge_out, ei2 ----
  rowmax_bin<<<1, 128, 0, stream>>>(ADJP, GN2, out_edge, COLS, out_ei2);

  // ---- SAGE 3 on xp (128 nodes x 128 feats -> 2), tanh ----
  sage3_agg<<<(GN2 * 128 + 255) / 256, 256, 0, stream>>>(XP, COLS, GN2, 128, AGG3, CNT3);
  sage_lin<<<(GN2 * 2 + 255) / 256, 256, 0, stream>>>(XP, AGG3, CNT3, s3wl, s3bl, s3wr, GN2, 128, 2, out_nodes);
  rownorm<<<1, 128, 0, stream>>>(out_nodes, GN2, 2, 1);

  // ---- diff_pool #2 reductions + loss scalars ----
  finalize<<<1, 1, 0, stream>>>(out_nodes, SCAL, out);
}
